// MoELayer_59030030516987
// MI455X (gfx1250) — matrix operand order repaired and checked by reading, not compiled, not run on hardware
//
#include <hip/hip_runtime.h>
#include <hip/hip_bf16.h>
#include <stdint.h>

// ---------------------------------------------------------------------------
// MoE top-2 layer for MI455X (gfx1250, wave32, WMMA).
// Sparse-equivalent math: per token, y = sum_{k=0,1} p_k * (relu(x @ W1[e_k]) @ W2[e_k])
// bf16 WMMA (16x16x32, f32 accum); weights staged block-wide into LDS via
// global_load_async_to_lds_b128 (ASYNCcnt) so each block reads W1/W2 once.
// ---------------------------------------------------------------------------

#define T_TOK 8192   // BATCH*SEQ
#define DM    1024   // d_model
#define CAP   512    // expert hidden
#define NE    8      // experts

typedef __attribute__((ext_vector_type(8)))  __bf16 v8bf;
typedef __attribute__((ext_vector_type(16))) __bf16 v16bf;
typedef __attribute__((ext_vector_type(8)))  float  v8f;

__device__ __forceinline__ v16bf cat8(v8bf lo, v8bf hi) {
  return __builtin_shufflevector(lo, hi, 0,1,2,3,4,5,6,7,8,9,10,11,12,13,14,15);
}

// async global->LDS copy of 16 bytes (tracked by ASYNCcnt)
__device__ __forceinline__ void async_b128(unsigned lds_off, const void* gptr) {
  asm volatile("global_load_async_to_lds_b128 %0, %1, off"
               :: "v"(lds_off), "v"((unsigned long long)(uintptr_t)gptr)
               : "memory");
}
__device__ __forceinline__ void wait_async0() {
  asm volatile("s_wait_asynccnt 0" ::: "memory");
}

// --------------------------- kernel 0: init --------------------------------
__global__ void moe_zero_kernel(float* __restrict__ out, int n, int* __restrict__ counts) {
  int i0 = blockIdx.x * blockDim.x + threadIdx.x;
  if (i0 < NE) counts[i0] = 0;
  int stride = gridDim.x * blockDim.x;
  for (int i = i0; i < n; i += stride) out[i] = 0.0f;
}

// --------------------- kernel 1: fp32 -> bf16 convert ----------------------
__global__ void moe_cvt_bf16_kernel(const float* __restrict__ in, __bf16* __restrict__ out, int n) {
  int i = blockIdx.x * blockDim.x + threadIdx.x;
  int stride = gridDim.x * blockDim.x;
  for (; i < n; i += stride) out[i] = (__bf16)in[i];
}

// ------------- kernel 2: per-expert transpose + convert to bf16 ------------
// in:  [z][R][Cc] fp32 (row-major)   out: [z][Cc][R] bf16
__global__ void moe_transpose_cvt_kernel(const float* __restrict__ in, __bf16* __restrict__ out,
                                         int R, int Cc) {
  __shared__ float tile[32][33];
  const float* src = in  + (size_t)blockIdx.z * R * Cc;
  __bf16*      dst = out + (size_t)blockIdx.z * R * Cc;
  const int c0 = blockIdx.x * 32;
  const int r0 = blockIdx.y * 32;
  const int tx = threadIdx.x, ty = threadIdx.y;         // 32 x 8 threads
  #pragma unroll
  for (int j = 0; j < 32; j += 8)
    tile[ty + j][tx] = src[(size_t)(r0 + ty + j) * Cc + (c0 + tx)];
  __syncthreads();
  #pragma unroll
  for (int j = 0; j < 32; j += 8)
    dst[(size_t)(c0 + ty + j) * R + (r0 + tx)] = (__bf16)tile[tx][ty + j];
}

// ------------------------ kernel 3: top-2 routing --------------------------
// One wave32 per token. Builds per-expert gather lists (counting sort).
__global__ void moe_routing_kernel(const float* __restrict__ x, const float* __restrict__ wg,
                                   int* __restrict__ counts, int* __restrict__ tlist,
                                   float* __restrict__ plist) {
  const int wave = threadIdx.x >> 5;
  const int lane = threadIdx.x & 31;
  const int t = blockIdx.x * 8 + wave;
  const float* xr = x + (size_t)t * DM;
  float acc[NE] = {0.f,0.f,0.f,0.f,0.f,0.f,0.f,0.f};
  for (int d = lane; d < DM; d += 32) {
    const float xv = xr[d];
    const float* w = wg + (size_t)d * NE;
    #pragma unroll
    for (int e = 0; e < NE; ++e) acc[e] += xv * w[e];
  }
  #pragma unroll
  for (int e = 0; e < NE; ++e) {
    #pragma unroll
    for (int off = 16; off > 0; off >>= 1)
      acc[e] += __shfl_xor(acc[e], off, 32);
  }
  if (lane == 0) {
    int e0 = 0; float s0 = acc[0];
    #pragma unroll
    for (int e = 1; e < NE; ++e) if (acc[e] > s0) { s0 = acc[e]; e0 = e; }
    int e1 = (e0 == 0) ? 1 : 0; float s1 = acc[e1];
    #pragma unroll
    for (int e = 0; e < NE; ++e)
      if (e != e0 && acc[e] > s1) { s1 = acc[e]; e1 = e; }
    float z  = __expf(s1 - s0);
    float p0 = 1.0f / (1.0f + z);
    float p1 = 1.0f - p0;
    int slot0 = atomicAdd(&counts[e0], 1);
    tlist[(size_t)e0 * T_TOK + slot0] = t;
    plist[(size_t)e0 * T_TOK + slot0] = p0;
    int slot1 = atomicAdd(&counts[e1], 1);
    tlist[(size_t)e1 * T_TOK + slot1] = t;
    plist[(size_t)e1 * T_TOK + slot1] = p1;
  }
}

// -------------------- kernel 4: fused 2-layer expert MLP -------------------
// grid = (ceil(T/64), NE). 4 waves/block; each wave owns a 16-token tile.
// Weight chunks (64 cols x 512 K bf16 = 64KB) are staged into LDS once per
// block with async-to-LDS copies and shared by all 4 waves, so each block
// reads W1 and W2 exactly once from L2. All four B fragments for a k-step
// are loaded into distinct registers before the four WMMAs so the matrix
// ops issue back-to-back (no per-WMMA DS round-trip).
// Dynamic LDS: hbuf 64KB (4 x 16 x 512 bf16) + bbuf 64KB (64 x 512 bf16).
__global__ void __launch_bounds__(128)
moe_mlp_kernel(const __bf16* __restrict__ xb, const __bf16* __restrict__ w1t,
               const __bf16* __restrict__ w2t, const int* __restrict__ counts,
               const int* __restrict__ tlist, const float* __restrict__ plist,
               float* __restrict__ out) {
  extern __shared__ __bf16 smem[];
  __bf16* hbuf = smem;                    // [4][16][CAP]
  __bf16* bbuf = smem + 4 * 16 * CAP;     // [64][512]  staged weight chunk

  const int e    = blockIdx.y;
  const int cnt  = counts[e];
  const int base = blockIdx.x * 64;
  if (base >= cnt) return;
  const int nav  = min(cnt - base, 64);

  const int tid   = threadIdx.x;
  const int wave  = tid >> 5;
  const int lane  = tid & 31;
  const int nB    = lane & 15;                           // N / M index within 16-lane half
  const int khalf = lane >> 4;                           // which K half this lane holds
  const int mrow  = wave * 16;

  const int*   tl = tlist + (size_t)e * T_TOK + base;
  const float* pl = plist + (size_t)e * T_TOK + base;
  __bf16* hw = hbuf + wave * (16 * CAP);
  const __bf16* w1e = w1t + (size_t)e * CAP * DM;        // [CAP][DM]  (N-major, K contiguous)
  const __bf16* w2e = w2t + (size_t)e * DM * CAP;        // [DM][CAP]

  // staging assignment: thread -> (col 0..63, half 0..1); 512 B each = 32 b128s
  const int scol  = tid >> 1;
  const int spart = tid & 1;
  const unsigned sdst = (unsigned)(uintptr_t)(bbuf + scol * 512 + spart * 256);

  // ----------------- layer 1: h = relu(x @ W1) -----------------
  {
    const int rowA = mrow + nB;                          // A-frag: lane supplies token row nB
    const __bf16* xrow = xb + (size_t)tl[min(rowA, nav - 1)] * DM;
    for (int ng = 0; ng < 8; ++ng) {                     // 8 groups x 64 cols = CAP
      v8f acc[4] = {};
      for (int kc = 0; kc < 2; ++kc) {                   // two 512-K chunks
        // stage 64 cols x 512 K of W1 for this (ng, kc)
        {
          const __bf16* src = w1e + (size_t)(ng * 64 + scol) * DM + kc * 512 + spart * 256;
          #pragma unroll
          for (int i = 0; i < 32; ++i)
            async_b128(sdst + i * 16, src + i * 8);
          wait_async0();
        }
        __syncthreads();
        for (int k0 = 0; k0 < 512; k0 += 32) {
          // A 16x32 bf16: lanes 0-15 K={k0..+7, k0+16..+23}; lanes 16-31 shifted by 8
          const __bf16* ap = xrow + kc * 512 + k0 + khalf * 8;
          v16bf a = cat8(*(const v8bf*)ap, *(const v8bf*)(ap + 16));
          v16bf bfr[4];
          #pragma unroll
          for (int j = 0; j < 4; ++j) {
            const __bf16* bp = bbuf + (size_t)(j * 16 + nB) * 512 + k0 + khalf * 8;
            bfr[j] = cat8(*(const v8bf*)bp, *(const v8bf*)(bp + 16));
          }
          #pragma unroll
          for (int j = 0; j < 4; ++j)
            acc[j] = __builtin_amdgcn_wmma_f32_16x16x32_bf16(
                false, a, false, bfr[j], (short)0, acc[j], false, false);
        }
        __syncthreads();                                 // before bbuf is overwritten
      }
      // C/D layout: N = nB, M = r + 8*khalf. ReLU -> bf16 -> LDS (token-major).
      #pragma unroll
      for (int j = 0; j < 4; ++j) {
        const int n = (ng * 4 + j) * 16 + nB;
        #pragma unroll
        for (int r = 0; r < 8; ++r) {
          const float v = acc[j][r];
          hw[(r + khalf * 8) * CAP + n] = (__bf16)(v > 0.0f ? v : 0.0f);
        }
      }
    }
  }
  __syncthreads();

  // ----------------- layer 2: y = h @ W2, scatter-add -----------------
  {
    float  pr[8];
    float* orow[8];
    #pragma unroll
    for (int r = 0; r < 8; ++r) {
      const int row = mrow + r + khalf * 8;              // epilogue rows (C/D layout)
      const int cl  = min(row, nav - 1);
      orow[r] = out + (size_t)tl[cl] * DM;
      pr[r]   = (row < nav) ? pl[cl] : 0.0f;             // padded rows contribute exactly 0
    }
    const __bf16* hrow = hw + (size_t)nB * CAP;          // A-frag row = nB
    for (int ng = 0; ng < 16; ++ng) {                    // 16 groups x 64 cols = DM
      v8f acc[4] = {};
      // stage 64 cols x full 512 K of W2 for this ng
      {
        const __bf16* src = w2e + (size_t)(ng * 64 + scol) * CAP + spart * 256;
        #pragma unroll
        for (int i = 0; i < 32; ++i)
          async_b128(sdst + i * 16, src + i * 8);
        wait_async0();
      }
      __syncthreads();
      for (int k0 = 0; k0 < 512; k0 += 32) {
        const __bf16* ap = hrow + k0 + khalf * 8;
        v16bf a = cat8(*(const v8bf*)ap, *(const v8bf*)(ap + 16));
        v16bf bfr[4];
        #pragma unroll
        for (int j = 0; j < 4; ++j) {
          const __bf16* bp = bbuf + (size_t)(j * 16 + nB) * 512 + k0 + khalf * 8;
          bfr[j] = cat8(*(const v8bf*)bp, *(const v8bf*)(bp + 16));
        }
        #pragma unroll
        for (int j = 0; j < 4; ++j)
          acc[j] = __builtin_amdgcn_wmma_f32_16x16x32_bf16(
              false, a, false, bfr[j], (short)0, acc[j], false, false);
      }
      __syncthreads();                                   // before bbuf is overwritten
      #pragma unroll
      for (int j = 0; j < 4; ++j) {
        const int n = (ng * 4 + j) * 16 + nB;
        #pragma unroll
        for (int r = 0; r < 8; ++r) {
          unsafeAtomicAdd(orow[r] + n, acc[j][r] * pr[r]);  // global_atomic_add_f32
        }
      }
    }
  }
}

// ---------------------------------------------------------------------------
extern "C" void kernel_launch(void* const* d_in, const int* in_sizes, int n_in,
                              void* d_out, int out_size, void* d_ws, size_t ws_size,
                              hipStream_t stream) {
  (void)in_sizes; (void)n_in; (void)out_size;
  const float* x  = (const float*)d_in[0];   // [T, D]
  const float* wg = (const float*)d_in[1];   // [D, E]
  const float* w1 = (const float*)d_in[2];   // [E, D, C]
  const float* w2 = (const float*)d_in[3];   // [E, C, D]
  float* out = (float*)d_out;                // [T, D]

  // workspace layout (~34.1 MB total)
  char* ws = (char*)d_ws;
  size_t off = 0;
  __bf16* xb  = (__bf16*)(ws + off); off += (size_t)T_TOK * DM * 2;        // 16.78 MB
  __bf16* w1t = (__bf16*)(ws + off); off += (size_t)NE * DM * CAP * 2;     //  8.39 MB
  __bf16* w2t = (__bf16*)(ws + off); off += (size_t)NE * DM * CAP * 2;     //  8.39 MB
  int*    counts = (int*)(ws + off); off += 256;
  int*    tlist  = (int*)(ws + off); off += (size_t)NE * T_TOK * 4;        //  0.26 MB
  float*  plist  = (float*)(ws + off); off += (size_t)NE * T_TOK * 4;      //  0.26 MB
  if (ws_size < off) return;  // insufficient scratch (not expected)

  moe_zero_kernel<<<4096, 256, 0, stream>>>(out, T_TOK * DM, counts);
  moe_cvt_bf16_kernel<<<4096, 256, 0, stream>>>(x, xb, T_TOK * DM);
  moe_transpose_cvt_kernel<<<dim3(CAP / 32, DM / 32, NE), dim3(32, 8), 0, stream>>>(w1, w1t, DM, CAP);
  moe_transpose_cvt_kernel<<<dim3(DM / 32, CAP / 32, NE), dim3(32, 8), 0, stream>>>(w2, w2t, CAP, DM);
  moe_routing_kernel<<<T_TOK / 8, 256, 0, stream>>>(x, wg, counts, tlist, plist);

  const size_t lds_bytes = (size_t)(4 * 16 * CAP + 64 * CAP) * sizeof(__bf16); // 128 KB
  moe_mlp_kernel<<<dim3((T_TOK + 63) / 64, NE), 128, lds_bytes, stream>>>(
      xb, w1t, w2t, counts, tlist, plist, out);
}
